// NeRF_MLP_Compose_MultiResHashEncoding_56315611185300
// MI455X (gfx1250) — matrix operand order repaired and checked by reading, not compiled, not run on hardware
//
#include <hip/hip_runtime.h>

typedef __attribute__((ext_vector_type(16))) _Float16 v16h;
typedef __attribute__((ext_vector_type(8)))  _Float16 v8h;
typedef __attribute__((ext_vector_type(8)))  float    v8f;

#define NLEV 16
#define TBITS 19
#define TSIZE (1 << TBITS)
#define HID 256
#define IN_F 32
#define NBLK 3
#define NCLS 4
#define ALIGN_ROWS 64
#define PX 272   // X-tile LDS pitch in f16 (row base 16B aligned)
#define PW 40    // W-panel LDS pitch in f16 (row base 16B aligned)
#define SCALE_F   65536.0f
#define INV_SCALE (1.0f / 65536.0f)
// per-class f16 weight blob: L0 [256n x 32k] + 3 x [256n x 256k], K contiguous
#define WT_L0    (HID * IN_F)
#define WT_LR    (HID * HID)
#define WT_CLS   (WT_L0 + NBLK * WT_LR)

__device__ __constant__ int c_res[NLEV] = {16, 20, 25, 32, 40, 50, 64, 80,
                                           101, 128, 161, 203, 256, 322, 406, 512};

// ---------------------------------------------------------------- utilities
__device__ inline v16h mk16(v8h lo, v8h hi) {
  v16h r;
#pragma unroll
  for (int i = 0; i < 8; ++i) { r[i] = lo[i]; r[i + 8] = hi[i]; }
  return r;
}

// ---------------------------------------------------------------- pass 0: zero counters
__global__ void k_zero16(int* p) {
  if (threadIdx.x < 16) p[threadIdx.x] = 0;
}

// ---------------------------------------------------------------- pass 1: per-class counts + map init
__global__ __launch_bounds__(256) void k_init_count(const int* __restrict__ lid,
                                                    int* __restrict__ counts,
                                                    int* __restrict__ map,
                                                    int B, int rowsCap) {
  int i = blockIdx.x * blockDim.x + threadIdx.x;
  if (i < rowsCap) map[i] = -1;
  int cls = (i < B) ? (lid[i] & 3) : -1;
  int lane = threadIdx.x & 31;
#pragma unroll
  for (int c = 0; c < 4; ++c) {
    unsigned m = (unsigned)__ballot(cls == c);   // wave32: low 32 bits
    if (cls == c && lane == (__ffs(m) - 1)) atomicAdd(&counts[c], __popc(m));
  }
}

// ---------------------------------------------------------------- pass 2: 64-aligned segment offsets
__global__ void k_offsets(int* wsI) {
  if (threadIdx.x == 0 && blockIdx.x == 0) {
    int off = 0;
#pragma unroll
    for (int c = 0; c < 4; ++c) {
      wsI[8 + c] = off;          // segment start
      wsI[4 + c] = off;          // running cursor
      int cnt = wsI[c];
      off += ((cnt + ALIGN_ROWS - 1) / ALIGN_ROWS) * ALIGN_ROWS;
    }
    wsI[12] = off;               // padded end
  }
}

// ---------------------------------------------------------------- pass 2b: one-time weight f32->f16 transpose
// Wt layout per class: [n][k] with k contiguous (L0 pitch 32, Lr pitch 256)
__global__ __launch_bounds__(256) void k_wconv(const float* __restrict__ W0,
                                               const float* __restrict__ Wres,
                                               _Float16* __restrict__ Wt) {
  int idx = blockIdx.x * blockDim.x + threadIdx.x;
  if (idx >= NCLS * WT_CLS) return;
  int c = idx / WT_CLS;
  int rem = idx % WT_CLS;
  float v;
  if (rem < WT_L0) {
    int n = rem / IN_F, k = rem % IN_F;
    v = W0[((size_t)c * IN_F + k) * HID + n];
  } else {
    int r = (rem - WT_L0) / WT_LR;
    int q = (rem - WT_L0) % WT_LR;
    int n = q / HID, k = q % HID;
    v = Wres[(((size_t)c * NBLK + r) * HID + k) * HID + n];
  }
  Wt[idx] = (_Float16)v;
}

// ---------------------------------------------------------------- pass 3: hash encode + class scatter
__global__ __launch_bounds__(256) void k_encode(const float* __restrict__ x,
                                                const int* __restrict__ lid,
                                                const float* __restrict__ emb,
                                                int* __restrict__ map,
                                                int* __restrict__ cursors,
                                                _Float16* __restrict__ feat,
                                                int B) {
  const unsigned PRIM[6] = {1u, 2654435761u, 805459861u, 3674653429u,
                            2097192037u, 1434869437u};
  int i = blockIdx.x * blockDim.x + threadIdx.x;
  int cls = (i < B) ? (lid[i] & 3) : -1;
  int lane = threadIdx.x & 31;
  int row = -1;
#pragma unroll
  for (int c = 0; c < 4; ++c) {                  // wave-aggregated slot assignment
    unsigned m = (unsigned)__ballot(cls == c);
    if (cls == c) {
      int leader = __ffs(m) - 1;
      int base = 0;
      if (lane == leader) base = atomicAdd(&cursors[c], __popc(m));
      base = __shfl(base, leader, 32);
      row = base + __popc(m & ((1u << lane) - 1u));
    }
  }
  if (i >= B) return;
  map[row] = i;

  float xi[6];
#pragma unroll
  for (int d = 0; d < 6; ++d) {
    float v = x[(size_t)i * 6 + d];
    xi[d] = fminf(fmaxf(v, 0.f), 1.f);
  }

  _Float16* fr = feat + (size_t)row * IN_F;
  for (int l = 0; l < NLEV; ++l) {
    float resf = (float)c_res[l];
    float grid = 1.f / resf;
    const float* embL = emb + (size_t)l * TSIZE * 2;

    unsigned h0 = 0;
    unsigned dlt[6];
    float w0a[6], w1a[6];
#pragma unroll
    for (int d = 0; d < 6; ++d) {
      float bf = floorf(xi[d] * resf);
      int bl = (int)bf;
      float vmin = bf * grid;
      float wd = (xi[d] - vmin) / (grid + 1e-6f);
      wd = fminf(fmaxf(wd, 0.f), 1.f);
      w1a[d] = wd;
      w0a[d] = 1.f - wd;
      unsigned u = (unsigned)bl * PRIM[d];
      h0 ^= u;
      dlt[d] = u ^ (u + PRIM[d]);   // XOR delta for bl -> bl+1 on dim d
    }

    // factor the 6-bit corner into two 3-bit halves.
    // hash: corner bit p selects dim (5-p)  -> low bits {5,4,3}, high bits {2,1,0}
    // weight: corner bit d selects dim d    -> low bits {0,1,2}, high bits {3,4,5}
    unsigned xl[8], xh[8];
    float wl[8], wh[8];
#pragma unroll
    for (int b = 0; b < 8; ++b) {
      unsigned hv = 0;
      if (b & 1) hv ^= dlt[5];
      if (b & 2) hv ^= dlt[4];
      if (b & 4) hv ^= dlt[3];
      xl[b] = hv;
      unsigned hv2 = 0;
      if (b & 1) hv2 ^= dlt[2];
      if (b & 2) hv2 ^= dlt[1];
      if (b & 4) hv2 ^= dlt[0];
      xh[b] = hv2;
      wl[b] = ((b & 1) ? w1a[0] : w0a[0]) * ((b & 2) ? w1a[1] : w0a[1]) *
              ((b & 4) ? w1a[2] : w0a[2]);
      wh[b] = ((b & 1) ? w1a[3] : w0a[3]) * ((b & 2) ? w1a[4] : w0a[4]) *
              ((b & 4) ? w1a[5] : w0a[5]);
    }

    float f0 = 0.f, f1 = 0.f;
#pragma unroll
    for (int hi = 0; hi < 8; ++hi) {
      unsigned hbase = h0 ^ xh[hi];
      float wb = wh[hi];
#pragma unroll
      for (int lo = 0; lo < 8; ++lo) {
        unsigned idx = (hbase ^ xl[lo]) & (TSIZE - 1);
        float wt = wb * wl[lo];
        const float2 ev = *(const float2*)(embL + 2 * (size_t)idx);
        f0 = fmaf(wt, ev.x, f0);
        f1 = fmaf(wt, ev.y, f1);
      }
    }
    fr[2 * l + 0] = (_Float16)(f0 * SCALE_F);
    fr[2 * l + 1] = (_Float16)(f1 * SCALE_F);
  }
}

// ---------------------------------------------------------------- pass 4: class-uniform WMMA MLP
// 64 rows per WG (4 row-tiles), 8 waves; wave w owns 2 N-tiles for ALL row-tiles.
// Single X buffer: activations rewritten in place after a post-k-loop barrier.
__global__ __launch_bounds__(256) void k_mlp(const _Float16* __restrict__ feat,
                                             const int* __restrict__ map,
                                             const int* __restrict__ wsI,
                                             const _Float16* __restrict__ Wt,
                                             const float* __restrict__ b0,
                                             const float* __restrict__ bres,
                                             const float* __restrict__ scales,
                                             const float* __restrict__ Wout,
                                             const float* __restrict__ bout,
                                             float* __restrict__ out) {
  __shared__ __align__(16) _Float16 sX[ALIGN_ROWS * PX];
  __shared__ __align__(16) _Float16 sW[HID * PW];
  __shared__ float sOut[ALIGN_ROWS];

  const int row0 = blockIdx.x * ALIGN_ROWS;
  if (row0 >= wsI[12]) return;                  // beyond padded classes (block-uniform)

  // class of this 64-row tile (segments are 64-aligned)
  int c = (row0 >= wsI[9]) + (row0 >= wsI[10]) + (row0 >= wsI[11]);

  const _Float16* WtC = Wt + (size_t)c * WT_CLS;
  const float* b0g   = b0 + (size_t)c * HID;
  const float* bresg = bres + (size_t)c * NBLK * HID;
  const float* scg   = scales + (size_t)c * NBLK;
  const float* Woutg = Wout + (size_t)c * HID;
  const float  boutc = bout[c];

  const int tid = threadIdx.x;
  const int wv = tid >> 5, lane = tid & 31;
  const int ntb = wv * 2;            // 2 N-tiles per wave: cols (ntb+t)*16
  const int lane_m = lane & 15, lg = lane >> 4;

  // stage initial 64x32 f16 features (16B per thread)
  {
    int r = tid >> 2, ch = tid & 3;
    const uint4* src = (const uint4*)(feat + (size_t)(row0 + r) * IN_F);
    *(uint4*)&sX[r * PX + ch * 8] = src[ch];
  }

  float hreg[4][2][8];               // fp32 residual state: [row-tile][n-tile][m]
  for (int L = 0; L < 4; ++L) {
    const _Float16* WtL;
    int K;
    if (L == 0) { WtL = WtC; K = IN_F; }
    else        { WtL = WtC + WT_L0 + (size_t)(L - 1) * WT_LR; K = HID; }

    v8f acc[4][2];
#pragma unroll
    for (int rt = 0; rt < 4; ++rt)
#pragma unroll
      for (int t = 0; t < 2; ++t) { v8f z = {}; acc[rt][t] = z; }

    const int panels = K / 32;
    for (int p = 0; p < panels; ++p) {
      __syncthreads();                          // sW reuse + in-place X writes
      {
        // async global->LDS copy of one 64B W^T row per thread (GVS mode:
        // inst offset is applied to BOTH global and LDS addresses, so the
        // 80B LDS pitch is preserved with a single base per side).
        unsigned ldsa = (unsigned)(size_t)&sW[tid * PW];
        unsigned goff = (unsigned)(tid * (K * 2));
        const char* sbase = (const char*)WtL + (size_t)p * 64;
        asm volatile(
            "global_load_async_to_lds_b128 %0, %1, %2 offset:0\n\t"
            "global_load_async_to_lds_b128 %0, %1, %2 offset:16\n\t"
            "global_load_async_to_lds_b128 %0, %1, %2 offset:32\n\t"
            "global_load_async_to_lds_b128 %0, %1, %2 offset:48"
            :: "v"(ldsa), "v"(goff), "s"(sbase)
            : "memory");
        asm volatile("s_wait_asynccnt 0x0" ::: "memory");
      }
      __syncthreads();

      // A fragments, one per row-tile (16x32 f16: m = lane&15, K-halves by lane group)
      v16h af[4];
#pragma unroll
      for (int rt = 0; rt < 4; ++rt) {
        const _Float16* arow = sX + (rt * 16 + lane_m) * PX + p * 32 + lg * 8;
        af[rt] = mk16(*(const v8h*)arow, *(const v8h*)(arow + 16));
      }
#pragma unroll
      for (int t = 0; t < 2; ++t) {
        // B fragment (32x16 f16: n = lane&15, lane group selects K octets 8lg and 16+8lg)
        const _Float16* brow = sW + ((ntb + t) * 16 + lane_m) * PW + lg * 8;
        v16h bf = mk16(*(const v8h*)brow, *(const v8h*)(brow + 16));
#pragma unroll
        for (int rt = 0; rt < 4; ++rt)
          acc[rt][t] = __builtin_amdgcn_wmma_f32_16x16x32_f16(
              false, af[rt], false, bf, (short)0, acc[rt][t], false, false);
      }
    }
    __syncthreads();                  // all X reads done -> safe to rewrite in place

    // epilogue: bias + relu (+ scaled residual), all fp32
    if (L == 0) {
#pragma unroll
      for (int t = 0; t < 2; ++t) {
        float bn = SCALE_F * b0g[(ntb + t) * 16 + lane_m];
#pragma unroll
        for (int rt = 0; rt < 4; ++rt)
#pragma unroll
          for (int j = 0; j < 8; ++j)
            hreg[rt][t][j] = fmaxf(acc[rt][t][j] + bn, 0.f);
      }
    } else {
      const float* brg = bresg + (size_t)(L - 1) * HID;
      float sc = scg[L - 1];
#pragma unroll
      for (int t = 0; t < 2; ++t) {
        float bn = SCALE_F * brg[(ntb + t) * 16 + lane_m];
#pragma unroll
        for (int rt = 0; rt < 4; ++rt)
#pragma unroll
          for (int j = 0; j < 8; ++j)
            hreg[rt][t][j] = sc * fmaxf(acc[rt][t][j] + bn, 0.f) + hreg[rt][t][j];
      }
    }

    if (L < 3) {                      // write next X in place (C-layout -> row major)
#pragma unroll
      for (int t = 0; t < 2; ++t) {
        int col = (ntb + t) * 16 + lane_m;
#pragma unroll
        for (int rt = 0; rt < 4; ++rt)
#pragma unroll
          for (int j = 0; j < 8; ++j)
            sX[(rt * 16 + lg * 8 + j) * PX + col] = (_Float16)hreg[rt][t][j];
      }
    }
  }

  // output head: out[m] = (sum_n h[m][n] * Wout[n]) / S + bout
  if (tid < ALIGN_ROWS) sOut[tid] = 0.f;
  __syncthreads();
  float wvv[2];
#pragma unroll
  for (int t = 0; t < 2; ++t) wvv[t] = Woutg[(ntb + t) * 16 + lane_m];
#pragma unroll
  for (int rt = 0; rt < 4; ++rt)
#pragma unroll
    for (int j = 0; j < 8; ++j) {
      float v = hreg[rt][0][j] * wvv[0] + hreg[rt][1][j] * wvv[1];
      atomicAdd(&sOut[rt * 16 + lg * 8 + j], v);
    }
  __syncthreads();
  if (tid < ALIGN_ROWS) {
    int p = map[row0 + tid];
    if (p >= 0) out[p] = sOut[tid] * INV_SCALE + boutc;
  }
}

// ---------------------------------------------------------------- launcher
extern "C" void kernel_launch(void* const* d_in, const int* in_sizes, int n_in,
                              void* d_out, int out_size, void* d_ws, size_t ws_size,
                              hipStream_t stream) {
  const float* x      = (const float*)d_in[0];
  const int*   lid    = (const int*)d_in[1];
  const float* emb    = (const float*)d_in[2];
  const float* W0     = (const float*)d_in[3];
  const float* b0     = (const float*)d_in[4];
  const float* Wres   = (const float*)d_in[5];
  const float* bres   = (const float*)d_in[6];
  const float* scales = (const float*)d_in[7];
  const float* Wout   = (const float*)d_in[8];
  const float* bout   = (const float*)d_in[9];
  float* out = (float*)d_out;

  int B = in_sizes[0] / 6;
  int rowsCap = ((B + ALIGN_ROWS - 1) / ALIGN_ROWS) * ALIGN_ROWS + 4 * ALIGN_ROWS;

  int* wsI = (int*)d_ws;           // [0..3]=counts [4..7]=cursors [8..12]=offsets
  int* map = wsI + 16;             // byte 64
  size_t featOff = (64 + (size_t)rowsCap * 4 + 255) & ~(size_t)255;
  size_t wtOff = (featOff + (size_t)rowsCap * IN_F * 2 + 255) & ~(size_t)255;
  _Float16* feat = (_Float16*)((char*)d_ws + featOff);
  _Float16* Wt   = (_Float16*)((char*)d_ws + wtOff);

  k_zero16<<<1, 32, 0, stream>>>(wsI);
  k_init_count<<<(rowsCap + 255) / 256, 256, 0, stream>>>(lid, wsI, map, B, rowsCap);
  k_offsets<<<1, 1, 0, stream>>>(wsI);
  k_wconv<<<(NCLS * WT_CLS + 255) / 256, 256, 0, stream>>>(W0, Wres, Wt);
  k_encode<<<(B + 255) / 256, 256, 0, stream>>>(x, lid, emb, map, wsI + 4, feat, B);
  k_mlp<<<rowsCap / ALIGN_ROWS, 256, 0, stream>>>(feat, map, wsI, Wt, b0, bres,
                                                  scales, Wout, bout, out);
}
